// PatchAttention_40278203302376
// MI455X (gfx1250) — matrix operand order repaired and checked by reading, not compiled, not run on hardware
//
#include <hip/hip_runtime.h>
#include <hip/hip_bf16.h>
#include <math.h>

typedef __attribute__((ext_vector_type(16))) _Float16 v16h;
typedef __attribute__((ext_vector_type(8)))  float    v8f;
typedef __attribute__((ext_vector_type(4)))  unsigned int v4u;
typedef __attribute__((ext_vector_type(8)))  int      v8i;
typedef __attribute__((ext_vector_type(4)))  int      v4i;

#define NTOK   512
#define DMODEL 256
#define NHEAD  8
#define HDIM   32
#define HIDF   682          // SwiGLU hidden
#define NP1    1376         // 2*HIDF padded to /16
#define KP2    704          // HIDF padded to /32
#define QK_SCALE 0.17677669529663689f   // 1/sqrt(32)

__device__ __forceinline__ float silu_f(float x) { return x / (1.0f + __expf(-x)); }

// ---- WMMA fragment helpers (CDNA5 16x16x32 f16, wave32) -------------------
__device__ __forceinline__ v16h frag_a16(const _Float16* A, int lda, int lane) {
    int r = lane & 15, kb = (lane >> 4) * 8;
    const _Float16* p = A + (long)r * lda;
    v16h f;
#pragma unroll
    for (int t = 0; t < 8; ++t) { f[t] = p[kb + t]; f[8 + t] = p[16 + kb + t]; }
    return f;
}
__device__ __forceinline__ v16h frag_a16_rows(const _Float16* A, int lda, int lane, int rows) {
    int r = lane & 15, kb = (lane >> 4) * 8;
    v16h f;
    if (r < rows) {
        const _Float16* p = A + (long)r * lda;
#pragma unroll
        for (int t = 0; t < 8; ++t) { f[t] = p[kb + t]; f[8 + t] = p[16 + kb + t]; }
    } else {
#pragma unroll
        for (int t = 0; t < 16; ++t) f[t] = (_Float16)0.0f;
    }
    return f;
}
__device__ __forceinline__ v16h frag_bT16(const _Float16* BT, int ldb, int lane) {
    int n = lane & 15, kb = (lane >> 4) * 8;
    const _Float16* p = BT + (long)n * ldb;
    v16h f;
#pragma unroll
    for (int t = 0; t < 8; ++t) { f[t] = p[kb + t]; f[8 + t] = p[16 + kb + t]; }
    return f;
}
__device__ __forceinline__ v8f wmma16(v16h a, v16h b, v8f c) {
    return __builtin_amdgcn_wmma_f32_16x16x32_f16(false, a, false, b, (short)0, c, false, false);
}
__device__ __forceinline__ v8f zero8() {
    v8f z;
#pragma unroll
    for (int t = 0; t < 8; ++t) z[t] = 0.0f;
    return z;
}

// ---- TDM: DMA a 2-D f16 tile (rows x cols, row stride = cols) into LDS ----
// D# built per CDNA5 ISA ch.8: group0 {count,lds,global,type}, group1 {dims}.
__device__ __forceinline__ void tdm_load_tile_f16(const _Float16* gsrc, void* lds_dst,
                                                  unsigned rows, unsigned cols) {
    unsigned lds_off = (unsigned)(uintptr_t)lds_dst;        // low 32b of flat = LDS offset
    unsigned long long ga = (unsigned long long)(uintptr_t)gsrc;
    v4u g0;
    g0[0] = 1u;                                             // count=1, user descriptor
    g0[1] = lds_off;                                        // lds_addr (bytes)
    g0[2] = (unsigned)ga;                                   // global_addr[31:0]
    g0[3] = (unsigned)((ga >> 32) & 0x01FFFFFFu) | (2u << 30);  // addr[56:32] | type=2
    v8i g1;
    g1[0] = (int)(1u << 16);                                // data_size=1 (2 bytes)
    g1[1] = (int)(cols << 16);                              // tensor_dim0[15:0] @bits63:48
    g1[2] = (int)((cols >> 16) & 0xFFFFu) | (int)(rows << 16);  // td0 hi | tensor_dim1 lo
    g1[3] = (int)((rows >> 16) & 0xFFFFu) | (int)(cols << 16);  // td1 hi | tile_dim0
    g1[4] = (int)rows;                                      // tile_dim1 | tile_dim2=0
    g1[5] = (int)cols;                                      // tensor_dim0_stride[31:0]
    g1[6] = 0;                                              // stride hi | dim1_stride lo
    g1[7] = 0;
    v4i gz; gz[0] = 0; gz[1] = 0; gz[2] = 0; gz[3] = 0;     // groups 2/3 unused (2-D)
#if __has_include(<hip/amd_detail/amd_gfx1250_TDM.h>)
    v8i g4z; 
#pragma unroll
    for (int t = 0; t < 8; ++t) g4z[t] = 0;
    __builtin_amdgcn_tensor_load_to_lds(g0, g1, gz, gz, g4z, 0);   // clang-23 6-arg
#else
    __builtin_amdgcn_tensor_load_to_lds(g0, g1, gz, gz, 0);        // ROCm 7.2 5-arg
#endif
    __builtin_amdgcn_s_wait_tensorcnt(0);
}

// ---- weight convert: dst[n][k] = src[k][n] (f32->f16, zero-padded) --------
__global__ void k_convert_wT(const float* __restrict__ src, _Float16* __restrict__ dst,
                             int src_r, int src_c, int dst_r, int dst_c) {
    int idx = blockIdx.x * blockDim.x + threadIdx.x;
    if (idx >= dst_r * dst_c) return;
    int n = idx / dst_c, k = idx % dst_c;
    float v = (n < src_c && k < src_r) ? src[k * src_c + n] : 0.0f;
    dst[idx] = (_Float16)v;
}

// ---- layernorm (one 256-thread block per row) -----------------------------
__global__ void k_layernorm(const float* __restrict__ x, const float* __restrict__ g,
                            const float* __restrict__ b, _Float16* __restrict__ out) {
    __shared__ float red[256];
    int row = blockIdx.x, tid = threadIdx.x;
    float v = x[row * DMODEL + tid];
    red[tid] = v; __syncthreads();
    for (int s = 128; s > 0; s >>= 1) { if (tid < s) red[tid] += red[tid + s]; __syncthreads(); }
    float mean = red[0] * (1.0f / DMODEL); __syncthreads();
    float d = v - mean;
    red[tid] = d * d; __syncthreads();
    for (int s = 128; s > 0; s >>= 1) { if (tid < s) red[tid] += red[tid + s]; __syncthreads(); }
    float var = red[0] * (1.0f / DMODEL);
    out[row * DMODEL + tid] = (_Float16)(d * rsqrtf(var + 1e-5f) * g[tid] + b[tid]);
}

// ---- QKV GEMM: h16(512x256) @ qkv_wT -> q16(scaled), k16, v16T ------------
__global__ void k_gemm_qkv(const _Float16* __restrict__ h16, const _Float16* __restrict__ wT,
                           const float* __restrict__ bias,
                           _Float16* __restrict__ q16, _Float16* __restrict__ k16,
                           _Float16* __restrict__ v16T) {
    int gwave = (blockIdx.x * blockDim.x + threadIdx.x) >> 5;
    int lane = threadIdx.x & 31;
    int mt = gwave / 48, nt = gwave % 48;               // 32 x 48 tiles
    if (mt >= 32) return;
    const _Float16* Ab = h16 + mt * 16 * DMODEL;
    const _Float16* Bb = wT + nt * 16 * DMODEL;
    v8f acc = zero8();
#pragma unroll
    for (int kk = 0; kk < DMODEL / 32; ++kk) {
        __builtin_prefetch(Bb + (kk + 1) * 32, 0, 1);
        acc = wmma16(frag_a16(Ab + kk * 32, DMODEL, lane),
                     frag_bT16(Bb + kk * 32, DMODEL, lane), acc);
    }
    int m0 = (lane >> 4) * 8, n = lane & 15;
    int ncol = nt * 16 + n;
    float bb = bias[ncol];
#pragma unroll
    for (int r = 0; r < 8; ++r) {
        int i = mt * 16 + m0 + r;
        float val = acc[r] + bb;
        if (ncol < 256)      q16[i * DMODEL + ncol] = (_Float16)(val * QK_SCALE);
        else if (ncol < 512) k16[i * DMODEL + (ncol - 256)] = (_Float16)val;
        else                 v16T[(ncol - 512) * NTOK + i] = (_Float16)val;
    }
}

// ---- rel-bias: RB[i,j,h] = silu(rel@rb_w1+b1) @ rb_w2 + b2 ----------------
__global__ void k_relbias(const float* __restrict__ coords,
                          const float* __restrict__ rb_w1, const float* __restrict__ rb_b1,
                          const _Float16* __restrict__ rbw2T, const float* __restrict__ rb_b2,
                          float* __restrict__ RB) {
    __shared__ _Float16 Tl[64 * 256];
    int tid = threadIdx.x;
    int pbase = blockIdx.x * 64;
    int i = pbase >> 9;                        // constant across block (64 | 512)
    float cx = coords[i * 3], cy = coords[i * 3 + 1], cz = coords[i * 3 + 2];
    float w1x = rb_w1[tid], w1y = rb_w1[256 + tid], w1z = rb_w1[512 + tid], b1 = rb_b1[tid];
    int jbase = pbase & 511;
    for (int s = 0; s < 64; ++s) {
        int j = jbase + s;
        float dx = cx - coords[j * 3], dy = cy - coords[j * 3 + 1], dz = cz - coords[j * 3 + 2];
        Tl[s * 256 + tid] = (_Float16)silu_f(dx * w1x + dy * w1y + dz * w1z + b1);
    }
    __syncthreads();
    int wave = tid >> 5, lane = tid & 31;
    if (wave < 4) {
        v8f acc = zero8();
#pragma unroll
        for (int kk = 0; kk < 8; ++kk)
            acc = wmma16(frag_a16(Tl + wave * 16 * 256 + kk * 32, 256, lane),
                         frag_bT16(rbw2T + kk * 32, 256, lane), acc);
        int m0 = (lane >> 4) * 8, n = lane & 15;
        if (n < NHEAD) {
            float b2 = rb_b2[n];
#pragma unroll
            for (int r = 0; r < 8; ++r) {
                int p = pbase + wave * 16 + m0 + r;
                RB[p * NHEAD + n] = acc[r] + b2;
            }
        }
    }
}

// ---- scores: S[h,i,j] = q.k (scale folded) + RB, via WMMA with C init -----
__global__ void k_scores(const _Float16* __restrict__ q16, const _Float16* __restrict__ k16,
                         const float* __restrict__ RB, float* __restrict__ S) {
    int gwave = (blockIdx.x * blockDim.x + threadIdx.x) >> 5;
    int lane = threadIdx.x & 31;
    int h = gwave >> 10, rem = gwave & 1023;
    int it = rem >> 5, jt = rem & 31;
    int m0 = (lane >> 4) * 8, n = lane & 15;
    int jglob = jt * 16 + n;
    v8f acc;
#pragma unroll
    for (int r = 0; r < 8; ++r)
        acc[r] = RB[((it * 16 + m0 + r) * NTOK + jglob) * NHEAD + h];
    acc = wmma16(frag_a16(q16 + it * 16 * DMODEL + h * HDIM, DMODEL, lane),
                 frag_bT16(k16 + jt * 16 * DMODEL + h * HDIM, DMODEL, lane), acc);
    float* Sh = S + (long)h * NTOK * NTOK;
#pragma unroll
    for (int r = 0; r < 8; ++r)
        Sh[(it * 16 + m0 + r) * NTOK + jglob] = acc[r];
}

// ---- softmax over j, emit f16 weights -------------------------------------
__global__ void k_softmax(const float* __restrict__ S, _Float16* __restrict__ W16) {
    __shared__ float red[256];
    int row = blockIdx.x, tid = threadIdx.x;
    const float* Sr = S + (long)row * NTOK;
    float v0 = Sr[tid], v1 = Sr[tid + 256];
    red[tid] = fmaxf(v0, v1); __syncthreads();
    for (int s = 128; s > 0; s >>= 1) { if (tid < s) red[tid] = fmaxf(red[tid], red[tid + s]); __syncthreads(); }
    float mx = red[0]; __syncthreads();
    float e0 = __expf(v0 - mx), e1 = __expf(v1 - mx);
    red[tid] = e0 + e1; __syncthreads();
    for (int s = 128; s > 0; s >>= 1) { if (tid < s) red[tid] += red[tid + s]; __syncthreads(); }
    float inv = 1.0f / red[0];
    W16[(long)row * NTOK + tid] = (_Float16)(e0 * inv);
    W16[(long)row * NTOK + tid + 256] = (_Float16)(e1 * inv);
}

// ---- AGG[i,h,c] = sum_j W[h,i,j] * silu(rel_ij@rv_w1+b1)[c]  --------------
__global__ void k_agg(const float* __restrict__ coords,
                      const float* __restrict__ rv_w1, const float* __restrict__ rv_b1,
                      const _Float16* __restrict__ W16, _Float16* __restrict__ AGG16) {
    __shared__ _Float16 Ut[256 * 40];   // [c][j_local], padded stride 40
    int i = blockIdx.x, tid = threadIdx.x;
    int wave = tid >> 5, lane = tid & 31;
    float cx = coords[i * 3], cy = coords[i * 3 + 1], cz = coords[i * 3 + 2];
    float w1x = rv_w1[tid], w1y = rv_w1[256 + tid], w1z = rv_w1[512 + tid], b1 = rv_b1[tid];
    v8f acc0 = zero8(), acc1 = zero8();
    int cb = wave * 32;                 // this wave's two c-tiles
    for (int kk = 0; kk < 16; ++kk) {
        int jb = kk * 32;
#pragma unroll
        for (int s = 0; s < 32; ++s) {
            int j = jb + s;
            float dx = cx - coords[j * 3], dy = cy - coords[j * 3 + 1], dz = cz - coords[j * 3 + 2];
            Ut[tid * 40 + s] = (_Float16)silu_f(dx * w1x + dy * w1y + dz * w1z + b1);
        }
        __syncthreads();
        v16h a = frag_a16_rows(W16 + (long)i * NTOK + jb, NTOK * NTOK, lane, NHEAD);
        acc0 = wmma16(a, frag_bT16(Ut + cb * 40, 40, lane), acc0);
        acc1 = wmma16(a, frag_bT16(Ut + (cb + 16) * 40, 40, lane), acc1);
        __syncthreads();
    }
    int n = lane & 15;
    if (lane < 16) {                    // rows m=0..7 (heads) live in lanes 0..15
#pragma unroll
        for (int r = 0; r < 8; ++r) {
            AGG16[i * 2048 + r * 256 + cb + n]      = (_Float16)acc0[r];
            AGG16[i * 2048 + r * 256 + cb + 16 + n] = (_Float16)acc1[r];
        }
    }
}

// ---- attention out: x1 = x + W@V + (AGG@rv_w2 + b2), two GEMMs, one C -----
__global__ void k_attnout(const _Float16* __restrict__ W16, const _Float16* __restrict__ v16T,
                          const _Float16* __restrict__ AGG16, const _Float16* __restrict__ rvw2T,
                          const float* __restrict__ rv_b2, const float* __restrict__ x,
                          float* __restrict__ x1) {
    int gwave = (blockIdx.x * blockDim.x + threadIdx.x) >> 5;
    int lane = threadIdx.x & 31;
    int h = gwave >> 6, rem = gwave & 63;
    int it = rem >> 1, sub = rem & 1;
    int nglob = h * HDIM + sub * 16;
    int m0 = (lane >> 4) * 8, n = lane & 15;
    v8f acc;
#pragma unroll
    for (int r = 0; r < 8; ++r)
        acc[r] = x[(it * 16 + m0 + r) * DMODEL + nglob + n] + rv_b2[nglob + n];
    const _Float16* Aw = W16 + (long)h * NTOK * NTOK + it * 16 * NTOK;
    const _Float16* Bv = v16T + nglob * NTOK;
#pragma unroll
    for (int kk = 0; kk < 16; ++kk)
        acc = wmma16(frag_a16(Aw + kk * 32, NTOK, lane),
                     frag_bT16(Bv + kk * 32, NTOK, lane), acc);
    const _Float16* Ag = AGG16 + it * 16 * 2048 + h * 256;
    const _Float16* Br = rvw2T + nglob * 256;
#pragma unroll
    for (int kk = 0; kk < 8; ++kk)
        acc = wmma16(frag_a16(Ag + kk * 32, 2048, lane),
                     frag_bT16(Br + kk * 32, 256, lane), acc);
#pragma unroll
    for (int r = 0; r < 8; ++r)
        x1[(it * 16 + m0 + r) * DMODEL + nglob + n] = acc[r];
}

// ---- FFN1: h2 @ ffn_in_w ; A-tile staged to LDS via Tensor Data Mover -----
// grid (32 mt, 11 nt-chunks); all 8 waves of a block share one 16x256 A tile.
__global__ void k_ffn1(const _Float16* __restrict__ h2, const _Float16* __restrict__ wT,
                       const float* __restrict__ bias, float* __restrict__ AB) {
    __shared__ _Float16 Asm[16 * DMODEL];   // 8 KB A tile
    int tid = threadIdx.x;
    int wave = tid >> 5, lane = tid & 31;
    int mt = blockIdx.x, ntc = blockIdx.y;
    if (wave == 0) {
        tdm_load_tile_f16(h2 + mt * 16 * DMODEL, (void*)Asm, 16, DMODEL);
    }
    __syncthreads();                         // publish DMA'd tile to all waves
    int nt = ntc * 8 + wave;
    if (nt < NP1 / 16) {
        const _Float16* Bb = wT + nt * 16 * DMODEL;
        v8f acc = zero8();
#pragma unroll
        for (int kk = 0; kk < 8; ++kk) {
            __builtin_prefetch(Bb + (kk + 1) * 32, 0, 1);
            acc = wmma16(frag_a16(Asm + kk * 32, DMODEL, lane),
                         frag_bT16(Bb + kk * 32, DMODEL, lane), acc);
        }
        int m0 = (lane >> 4) * 8, n = lane & 15;
        int ncol = nt * 16 + n;
        float bb = (ncol < 2 * HIDF) ? bias[ncol] : 0.0f;
#pragma unroll
        for (int r = 0; r < 8; ++r)
            AB[(mt * 16 + m0 + r) * NP1 + ncol] = acc[r] + bb;
    }
}

// ---- SwiGLU gate: G16 = a * silu(b), K padded to 704 ----------------------
__global__ void k_gate(const float* __restrict__ AB, _Float16* __restrict__ G16) {
    int idx = blockIdx.x * blockDim.x + threadIdx.x;
    if (idx >= NTOK * KP2) return;
    int i = idx / KP2, k = idx % KP2;
    float g = 0.0f;
    if (k < HIDF) {
        float a = AB[i * NP1 + k], bg = AB[i * NP1 + HIDF + k];
        g = a * silu_f(bg);
    }
    G16[idx] = (_Float16)g;
}

// ---- FFN2: G @ ffn_out_w + bias + residual --------------------------------
__global__ void k_ffn2(const _Float16* __restrict__ G16, const _Float16* __restrict__ wT,
                       const float* __restrict__ bias, const float* __restrict__ x1,
                       float* __restrict__ out) {
    int gwave = (blockIdx.x * blockDim.x + threadIdx.x) >> 5;
    int lane = threadIdx.x & 31;
    int mt = gwave >> 4, nt = gwave & 15;
    const _Float16* Ab = G16 + mt * 16 * KP2;
    const _Float16* Bb = wT + nt * 16 * KP2;
    v8f acc = zero8();
#pragma unroll
    for (int kk = 0; kk < KP2 / 32; ++kk)
        acc = wmma16(frag_a16(Ab + kk * 32, KP2, lane),
                     frag_bT16(Bb + kk * 32, KP2, lane), acc);
    int m0 = (lane >> 4) * 8, n = lane & 15;
    int ncol = nt * 16 + n;
    float bb = bias[ncol];
#pragma unroll
    for (int r = 0; r < 8; ++r) {
        int i = mt * 16 + m0 + r;
        out[i * DMODEL + ncol] = x1[i * DMODEL + ncol] + acc[r] + bb;
    }
}

extern "C" void kernel_launch(void* const* d_in, const int* in_sizes, int n_in,
                              void* d_out, int out_size, void* d_ws, size_t ws_size,
                              hipStream_t stream) {
    const float* x       = (const float*)d_in[0];
    const float* coords  = (const float*)d_in[1];
    const float* ln1_g   = (const float*)d_in[2];
    const float* ln1_b   = (const float*)d_in[3];
    const float* qkv_w   = (const float*)d_in[4];
    const float* qkv_b   = (const float*)d_in[5];
    const float* rb_w1   = (const float*)d_in[6];
    const float* rb_b1   = (const float*)d_in[7];
    const float* rb_w2   = (const float*)d_in[8];
    const float* rb_b2   = (const float*)d_in[9];
    const float* rv_w1   = (const float*)d_in[10];
    const float* rv_b1   = (const float*)d_in[11];
    const float* rv_w2   = (const float*)d_in[12];
    const float* rv_b2   = (const float*)d_in[13];
    const float* ln2_g   = (const float*)d_in[14];
    const float* ln2_b   = (const float*)d_in[15];
    const float* ffn_in_w  = (const float*)d_in[16];
    const float* ffn_in_b  = (const float*)d_in[17];
    const float* ffn_out_w = (const float*)d_in[18];
    const float* ffn_out_b = (const float*)d_in[19];
    float* out = (float*)d_out;

    char* base = (char*)d_ws;
    size_t off = 0;
    auto alloc = [&](size_t bytes) -> char* {
        off = (off + 255) & ~(size_t)255;
        char* p = base + off; off += bytes; return p;
    };
    _Float16* h16     = (_Float16*)alloc(NTOK * DMODEL * 2);
    _Float16* qkvwT   = (_Float16*)alloc(768 * 256 * 2);
    _Float16* rbw2T   = (_Float16*)alloc(16 * 256 * 2);
    _Float16* rvw2T   = (_Float16*)alloc(256 * 256 * 2);
    _Float16* ffnwT   = (_Float16*)alloc((size_t)NP1 * 256 * 2);
    _Float16* ffn2wT  = (_Float16*)alloc((size_t)256 * KP2 * 2);
    _Float16* q16     = (_Float16*)alloc(NTOK * DMODEL * 2);
    _Float16* k16     = (_Float16*)alloc(NTOK * DMODEL * 2);
    _Float16* v16T    = (_Float16*)alloc(NTOK * DMODEL * 2);
    float*    RB      = (float*)alloc((size_t)NTOK * NTOK * NHEAD * 4);
    float*    S       = (float*)alloc((size_t)NHEAD * NTOK * NTOK * 4);
    _Float16* W16     = (_Float16*)alloc((size_t)NHEAD * NTOK * NTOK * 2);
    _Float16* AGG16   = (_Float16*)alloc((size_t)NTOK * 2048 * 2);
    float*    x1      = (float*)alloc(NTOK * DMODEL * 4);
    _Float16* h2      = (_Float16*)alloc(NTOK * DMODEL * 2);
    float*    AB      = (float*)alloc((size_t)NTOK * NP1 * 4);
    _Float16* G16     = (_Float16*)alloc((size_t)NTOK * KP2 * 2);

    // weight preprocessing (transpose + f16 + pad)
    k_convert_wT<<<(768 * 256 + 255) / 256, 256, 0, stream>>>(qkv_w, qkvwT, 256, 768, 768, 256);
    k_convert_wT<<<(16 * 256 + 255) / 256, 256, 0, stream>>>(rb_w2, rbw2T, 256, 8, 16, 256);
    k_convert_wT<<<(256 * 256 + 255) / 256, 256, 0, stream>>>(rv_w2, rvw2T, 256, 256, 256, 256);
    k_convert_wT<<<(NP1 * 256 + 255) / 256, 256, 0, stream>>>(ffn_in_w, ffnwT, 256, 2 * HIDF, NP1, 256);
    k_convert_wT<<<(256 * KP2 + 255) / 256, 256, 0, stream>>>(ffn_out_w, ffn2wT, HIDF, 256, 256, KP2);

    // attention branch
    k_layernorm<<<NTOK, 256, 0, stream>>>(x, ln1_g, ln1_b, h16);
    k_gemm_qkv<<<192, 256, 0, stream>>>(h16, qkvwT, qkv_b, q16, k16, v16T);
    k_relbias<<<(NTOK * NTOK) / 64, 256, 0, stream>>>(coords, rb_w1, rb_b1, rbw2T, rb_b2, RB);
    k_scores<<<1024, 256, 0, stream>>>(q16, k16, RB, S);
    k_softmax<<<NHEAD * NTOK, 256, 0, stream>>>(S, W16);
    k_agg<<<NTOK, 256, 0, stream>>>(coords, rv_w1, rv_b1, W16, AGG16);
    k_attnout<<<64, 256, 0, stream>>>(W16, v16T, AGG16, rvw2T, rv_b2, x, x1);

    // FFN branch
    k_layernorm<<<NTOK, 256, 0, stream>>>(x1, ln2_g, ln2_b, h2);
    k_ffn1<<<dim3(32, 11), 256, 0, stream>>>(h2, ffnwT, ffn_in_b, AB);
    k_gate<<<(NTOK * KP2 + 255) / 256, 256, 0, stream>>>(AB, G16);
    k_ffn2<<<64, 256, 0, stream>>>(G16, ffn2wT, ffn_out_b, x1, out);
}
